// BiLSTMModel_76149770158482
// MI455X (gfx1250) — matrix operand order repaired and checked by reading, not compiled, not run on hardware
//
#include <hip/hip_runtime.h>
#include <hip/hip_bf16.h>
#include <math.h>

typedef __attribute__((ext_vector_type(16))) __bf16 v16bf;
typedef __attribute__((ext_vector_type(8)))  __bf16 v8bf;
typedef __attribute__((ext_vector_type(8)))  float  v8f;

#define DEVI __device__ __forceinline__

DEVI unsigned short f2bf(float x) {           // round-to-nearest-even f32->bf16
  unsigned u = __float_as_uint(x);
  u += 0x7fffu + ((u >> 16) & 1u);
  return (unsigned short)(u >> 16);
}
DEVI float bflo(unsigned u) { return __uint_as_float(u << 16); }
DEVI float bfhi(unsigned u) { return __uint_as_float(u & 0xffff0000u); }
DEVI float sigf(float x)    { return 1.0f / (1.0f + expf(-x)); }

// ---------------- f32 -> bf16 strided converter ----------------
__global__ void k_cvt_bf16(unsigned short* __restrict__ dst, int dld,
                           const float* __restrict__ src, int sld,
                           int rows, int cols) {
  int idx = blockIdx.x * blockDim.x + threadIdx.x;
  if (idx >= rows * cols) return;
  int r = idx / cols, c = idx - r * cols;
  dst[(size_t)r * dld + c] = f2bf(src[(size_t)r * sld + c]);
}

__global__ void k_zero_us(unsigned short* p, int n) {
  int i = blockIdx.x * blockDim.x + threadIdx.x;
  if (i < n) p[i] = 0;
}

// ---------------- WMMA bf16 GEMM:  C[M,N] = A[M,K] @ B[N,K]^T (+bias)(+tanh) ---------
// grid = (N/64, M/256), block = 256 (8 waves); each wave computes a 32x64 tile.
// B (weights) is staged into LDS with CDNA5 async loads (ASYNCcnt), double-buffered in
// 64-wide K chunks; all 8 waves share the staged tile via ds_load_b128.
#define BPAD 72   // bf16 row stride in LDS: 144B keeps b128 aligned, 16 lanes conflict-free
__global__ __launch_bounds__(256)
void k_gemm_bf16(const unsigned short* __restrict__ A, int lda,
                 const unsigned short* __restrict__ B, int ldb,
                 float* __restrict__ Cf, unsigned short* __restrict__ Cb, int ldc,
                 const float* __restrict__ bias,
                 int K, int act, int nvalid) {
  __shared__ unsigned short sB[2][64 * BPAD];
  const int tid  = threadIdx.x;
  const int lane = tid & 31;
  const int wave = tid >> 5;
  const int lr   = lane & 15;
  const int hi   = lane >> 4;
  const int m0   = blockIdx.y * 256 + wave * 32;
  const int n0   = blockIdx.x * 64;

  // staging assignment: 256 threads cover 64 rows x 64 k-elements (32B per thread)
  const int srow = tid >> 2, sseg = tid & 3;
  const unsigned short* Bsrc = B + (size_t)(n0 + srow) * ldb + sseg * 16;
  const unsigned       sdst  = (unsigned)(size_t)&sB[0][srow * BPAD + sseg * 16];
  const unsigned       sbufB = (unsigned)(2 * 64 * BPAD); // byte distance buf0 -> buf1

  const unsigned short* Arow0 = A + (size_t)(m0 + lr) * lda;
  const unsigned short* Arow1 = A + (size_t)(m0 + 16 + lr) * lda;

  v8f acc[8];
#pragma unroll
  for (int i = 0; i < 8; ++i) acc[i] = (v8f){};

#define STAGE(SEL, KC)                                                              \
  do {                                                                              \
    unsigned _l = sdst + (unsigned)(SEL) * sbufB;                                   \
    unsigned long long _g = (unsigned long long)(size_t)(Bsrc + (KC));              \
    asm volatile("global_load_async_to_lds_b128 %0, %1, off"                        \
                 :: "v"(_l), "v"(_g) : "memory");                                   \
    asm volatile("global_load_async_to_lds_b128 %0, %1, off offset:16"              \
                 :: "v"(_l), "v"(_g) : "memory");                                   \
  } while (0)

  union BF { v16bf v; v8bf h[2]; };

  STAGE(0, 0);
  const int nchunks = K >> 6;
  for (int ch = 0; ch < nchunks; ++ch) {
    const int kc  = ch << 6;
    const int sel = ch & 1;
    asm volatile("s_wait_asynccnt 0x0" ::: "memory");  // my stage of buf[sel] done
    __syncthreads();                                   // everyone's stage visible

#pragma unroll
    for (int ks = 0; ks < 2; ++ks) {
      const int ka = kc + ks * 32;
      // A 16x32 fragments (global): lanes 0-15 K [0,16); lanes 16-31 K [16,32)
      BF a0, a1;
      a0.h[0] = *(const v8bf*)(Arow0 + ka + hi * 16);
      a0.h[1] = *(const v8bf*)(Arow0 + ka + hi * 16 + 8);
      a1.h[0] = *(const v8bf*)(Arow1 + ka + hi * 16);
      a1.h[1] = *(const v8bf*)(Arow1 + ka + hi * 16 + 8);
      // B 32x16 fragments from LDS: lane = one W row (B column), 16 contiguous K
      BF b[4];
#pragma unroll
      for (int nt = 0; nt < 4; ++nt) {
        const int o = (nt * 16 + lr) * BPAD + ks * 32 + hi * 16;
        b[nt].h[0] = *(const v8bf*)&sB[sel][o];
        b[nt].h[1] = *(const v8bf*)&sB[sel][o + 8];
      }
#pragma unroll
      for (int nt = 0; nt < 4; ++nt) {
        acc[nt]     = __builtin_amdgcn_wmma_f32_16x16x32_bf16(false, a0.v, false, b[nt].v,
                                                              (short)0, acc[nt],     false, false);
        acc[4 + nt] = __builtin_amdgcn_wmma_f32_16x16x32_bf16(false, a1.v, false, b[nt].v,
                                                              (short)0, acc[4 + nt], false, false);
      }
    }
    __syncthreads();                                   // all waves done reading buf[sel]
    if (ch + 1 < nchunks) STAGE(sel ^ 1, kc + 64);     // refill other buffer
  }
#undef STAGE

#pragma unroll
  for (int mt = 0; mt < 2; ++mt) {
    const int rowb = m0 + mt * 16 + hi * 8;
#pragma unroll
    for (int nt = 0; nt < 4; ++nt) {
      const int col = n0 + nt * 16 + lr;
      if (col < nvalid) {
        const float bv = bias ? bias[col] : 0.0f;
#pragma unroll
        for (int r = 0; r < 8; ++r) {
          float v = acc[mt * 4 + nt][r] + bv;
          if (act == 1) v = tanhf(v);
          size_t o = (size_t)(rowb + r) * (size_t)ldc + col;
          if (Cf) Cf[o] = v;
          if (Cb) Cb[o] = f2bf(v);
        }
      }
    }
  }
}

// ---------------- char LSTM init: Abuf = [emb(t=0) | h=0], c=0 ----------------
__global__ void k_char_init(unsigned short* __restrict__ Abuf, float* __restrict__ Cc,
                            const int* __restrict__ chars, const float* __restrict__ table) {
  int idx = blockIdx.x * blockDim.x + threadIdx.x; // 4096*256 threads
  int w = idx >> 8, k = idx & 255;
  Abuf[(size_t)w * 512 + 256 + k] = 0;                 // h0 = 0 (bf16)
  Cc[(size_t)w * 256 + k] = 0.0f;                      // c0 = 0
  int ch = chars[w * 16];
  Abuf[(size_t)w * 512 + k] = f2bf(table[(size_t)ch * 256 + k]);
}

// ---------------- char LSTM gate update (after gate GEMM) ----------------
__global__ void k_char_step(const float* __restrict__ G, unsigned short* __restrict__ Abuf,
                            float* __restrict__ Cc, unsigned short* __restrict__ wemb,
                            const int* __restrict__ chars, const int* __restrict__ lens,
                            const float* __restrict__ table, int t) {
  int idx = blockIdx.x * blockDim.x + threadIdx.x;
  int w = idx >> 8, k = idx & 255;
  const float* g = G + (size_t)w * 1024;
  float gi = g[k], gf = g[256 + k], gg = g[512 + k], go = g[768 + k];
  float c = Cc[(size_t)w * 256 + k];
  c = sigf(gf) * c + sigf(gi) * tanhf(gg);
  float h = sigf(go) * tanhf(c);
  Cc[(size_t)w * 256 + k] = c;
  Abuf[(size_t)w * 512 + 256 + k] = f2bf(h);           // h for next step's GEMM
  if (t + 1 < 16) {                                    // gather next char embedding
    int ch = chars[w * 16 + t + 1];
    Abuf[(size_t)w * 512 + k] = f2bf(table[(size_t)ch * 256 + k]);
  }
  int Lw = lens[w]; if (Lw < 1) Lw = 1;
  if (t == Lw - 1) wemb[(size_t)w * 256 + k] = f2bf(h);
}

// ---------------- word-level LSTM recurrence (batch 1, 4096 sequential steps) --------
// grid = 2 blocks (dir 0 = fwd, dir 1 = bwd), 1024 threads each.
// Thread j computes gate outputs n=j and n=j+1024; h lives in LDS; W_hh is bf16-packed.
__global__ __launch_bounds__(1024)
void k_lstm_seq(const float* __restrict__ P, const unsigned short* __restrict__ Whh,
                unsigned short* __restrict__ out, int T) {
  const int dir = blockIdx.x;
  const int j = threadIdx.x;
  __shared__ float h_sh[512];
  __shared__ float g_sh[2048];
  float c = 0.0f;
  if (j < 512) h_sh[j] = 0.0f;
  __syncthreads();
  const unsigned int* W0 = (const unsigned int*)(Whh + ((size_t)dir * 2048 + j) * 512);
  const unsigned int* W1 = (const unsigned int*)(Whh + ((size_t)dir * 2048 + j + 1024) * 512);
  const float* Pd = P + (size_t)dir * (size_t)T * 2048;
  for (int s = 0; s < T; ++s) {
    int t = dir ? (T - 1 - s) : s;
    const float* Pr = Pd + (size_t)t * 2048;
    float a0 = Pr[j], a1 = Pr[j + 1024];
#pragma unroll 8
    for (int kk = 0; kk < 256; ++kk) {
      float2 hv = *(const float2*)&h_sh[kk * 2];
      unsigned u0 = W0[kk], u1 = W1[kk];
      a0 = fmaf(bflo(u0), hv.x, a0); a0 = fmaf(bfhi(u0), hv.y, a0);
      a1 = fmaf(bflo(u1), hv.x, a1); a1 = fmaf(bfhi(u1), hv.y, a1);
    }
    g_sh[j] = a0; g_sh[j + 1024] = a1;
    __syncthreads();
    if (j < 512) {
      float gi = g_sh[j], gf = g_sh[j + 512], gg = g_sh[j + 1024], go = g_sh[j + 1536];
      c = sigf(gf) * c + sigf(gi) * tanhf(gg);
      float h = sigf(go) * tanhf(c);
      h_sh[j] = h;
      out[(size_t)t * 1024 + dir * 512 + j] = f2bf(h);
    }
    __syncthreads();
  }
}

extern "C" void kernel_launch(void* const* d_in, const int* in_sizes, int n_in,
                              void* d_out, int out_size, void* d_ws, size_t ws_size,
                              hipStream_t stream) {
  (void)in_sizes; (void)n_in; (void)out_size; (void)ws_size;
  const int*   chars = (const int*)  d_in[0];
  const int*   lens  = (const int*)  d_in[1];
  const float* table = (const float*)d_in[2];
  const float* cW_ih = (const float*)d_in[3];
  const float* cW_hh = (const float*)d_in[4];
  const float* cb    = (const float*)d_in[5];
  const float* W_ih0 = (const float*)d_in[6];
  const float* W_hh0 = (const float*)d_in[7];
  const float* b0    = (const float*)d_in[8];
  const float* W_ih1 = (const float*)d_in[9];
  const float* W_hh1 = (const float*)d_in[10];
  const float* b1    = (const float*)d_in[11];
  const float* fc1_w = (const float*)d_in[12];
  const float* fc1_b = (const float*)d_in[13];
  const float* fc2_w = (const float*)d_in[14];
  const float* fc2_b = (const float*)d_in[15];
  float* out = (float*)d_out;

  char* base = (char*)d_ws;
  size_t off = 0;
  auto alloc = [&](size_t bytes) -> char* {
    char* p = base + off;
    off = (off + bytes + 255) & ~(size_t)255;
    return p;
  };
  // persistent bf16 weights / activations
  unsigned short* Wc    = (unsigned short*)alloc((size_t)1024 * 512 * 2);   // [Wih|Whh] char
  unsigned short* Wih0b = (unsigned short*)alloc((size_t)2 * 2048 * 256 * 2);
  unsigned short* Whh0b = (unsigned short*)alloc((size_t)2 * 2048 * 512 * 2);
  unsigned short* Wih1b = (unsigned short*)alloc((size_t)2 * 2048 * 1024 * 2);
  unsigned short* Whh1b = (unsigned short*)alloc((size_t)2 * 2048 * 512 * 2);
  unsigned short* fc1wb = (unsigned short*)alloc((size_t)512 * 1024 * 2);
  unsigned short* fc2wb = (unsigned short*)alloc((size_t)64 * 512 * 2);     // padded 50->64
  unsigned short* wemb  = (unsigned short*)alloc((size_t)4096 * 256 * 2);
  unsigned short* out01 = (unsigned short*)alloc((size_t)4096 * 1024 * 2);
  unsigned short* out12 = (unsigned short*)alloc((size_t)4096 * 1024 * 2);
  unsigned short* h1bf  = (unsigned short*)alloc((size_t)4096 * 512 * 2);
  // time-shared 64MB region: char-phase {Abuf,G,Cc} overlaps word-phase P
  char* region = alloc((size_t)2 * 4096 * 2048 * 4);
  float*          P    = (float*)region;
  unsigned short* Abuf = (unsigned short*)region;                     // 4MB
  float*          G    = (float*)(region + (size_t)4  * 1024 * 1024); // 16MB
  float*          Cc   = (float*)(region + (size_t)20 * 1024 * 1024); // 4MB

  const int thr = 256;
#define G1(n) dim3((unsigned)(((n) + thr - 1) / thr))
  // weight conversions (fp32 -> bf16)
  k_cvt_bf16<<<G1(1024 * 256),  thr, 0, stream>>>(Wc,        512,  cW_ih, 256,  1024, 256);
  k_cvt_bf16<<<G1(1024 * 256),  thr, 0, stream>>>(Wc + 256,  512,  cW_hh, 256,  1024, 256);
  k_cvt_bf16<<<G1(4096 * 256),  thr, 0, stream>>>(Wih0b,     256,  W_ih0, 256,  4096, 256);
  k_cvt_bf16<<<G1(4096 * 512),  thr, 0, stream>>>(Whh0b,     512,  W_hh0, 512,  4096, 512);
  k_cvt_bf16<<<G1(4096 * 1024), thr, 0, stream>>>(Wih1b,     1024, W_ih1, 1024, 4096, 1024);
  k_cvt_bf16<<<G1(4096 * 512),  thr, 0, stream>>>(Whh1b,     512,  W_hh1, 512,  4096, 512);
  k_cvt_bf16<<<G1(512 * 1024),  thr, 0, stream>>>(fc1wb,     1024, fc1_w, 1024, 512,  1024);
  k_zero_us <<<G1(64 * 512),    thr, 0, stream>>>(fc2wb, 64 * 512);
  k_cvt_bf16<<<G1(50 * 512),    thr, 0, stream>>>(fc2wb,     512,  fc2_w, 512,  50,   512);

  // ---- char LSTM: 16 fused steps of [4096,512]@[512,1024]^T (WMMA) + gate update ----
  k_char_init<<<4096, 256, 0, stream>>>(Abuf, Cc, chars, table);
  for (int t = 0; t < 16; ++t) {
    k_gemm_bf16<<<dim3(1024 / 64, 4096 / 256), 256, 0, stream>>>(
        Abuf, 512, Wc, 512, G, nullptr, 1024, cb, 512, 0, 1024);
    k_char_step<<<4096, 256, 0, stream>>>(G, Abuf, Cc, wemb, chars, lens, table, t);
  }

  // ---- layer 0: batched input projections (WMMA) then sequential recurrence ----
  for (int d = 0; d < 2; ++d)
    k_gemm_bf16<<<dim3(2048 / 64, 4096 / 256), 256, 0, stream>>>(
        wemb, 256, Wih0b + (size_t)d * 2048 * 256, 256,
        P + (size_t)d * 4096 * 2048, nullptr, 2048, b0 + (size_t)d * 2048, 256, 0, 2048);
  k_lstm_seq<<<2, 1024, 0, stream>>>(P, Whh0b, out01, 4096);

  // ---- layer 1 ----
  for (int d = 0; d < 2; ++d)
    k_gemm_bf16<<<dim3(2048 / 64, 4096 / 256), 256, 0, stream>>>(
        out01, 1024, Wih1b + (size_t)d * 2048 * 1024, 1024,
        P + (size_t)d * 4096 * 2048, nullptr, 2048, b1 + (size_t)d * 2048, 1024, 0, 2048);
  k_lstm_seq<<<2, 1024, 0, stream>>>(P, Whh1b, out12, 4096);

  // ---- FC head: fc1 (tanh fused, bf16 out) then fc2 (N padded to 64, store cols<50) --
  k_gemm_bf16<<<dim3(512 / 64, 4096 / 256), 256, 0, stream>>>(
      out12, 1024, fc1wb, 1024, nullptr, h1bf, 512, fc1_b, 1024, 1, 512);
  k_gemm_bf16<<<dim3(1, 4096 / 256), 256, 0, stream>>>(
      h1bf, 512, fc2wb, 512, out, nullptr, 50, fc2_b, 512, 0, 50);
#undef G1
}
